// Model_4526895530596
// MI455X (gfx1250) — compile-verified, matrix operand order checked by reading, not run on hardware
//
#include <hip/hip_runtime.h>
#include <hip/hip_bf16.h>
#include <math.h>

// ---------------------------------------------------------------------------
// 2-layer LSTM (T=8192, F=256, H=1024, batch=1) + linear head for MI455X.
//   pack : f32 operand -> fragment-major (hi,lo) bf16 pair (bf16x3 split)
//   gemm : gx = X @ Wih^T + (bih+bhh), pure WMMA + coalesced b128 loads
//   scan : persistent-LSTM, Whh resident in ~258KB LDS/WGP, float4 LDS dots
//   head : out = y1 @ Wlin^T + blin
// ---------------------------------------------------------------------------

typedef __attribute__((ext_vector_type(16))) __bf16 v16bf;
typedef __attribute__((ext_vector_type(8)))  float  v8f;

#define Tseq   8192
#define Fdim   256
#define Hdim   1024
#define GATES  4096
#define NBLK   64          // persistent blocks for the scan (1 per WGP)
#define SROW   1032        // padded LDS row stride (floats), 16B-aligned rows

__device__ __forceinline__ float sigf(float x) {
  return 1.0f / (1.0f + __expf(-x));
}

// ---------------------------------------------------------------------------
// Device-wide generation barrier (all NBLK blocks co-resident: ~258KB LDS
// per block forces 1 block/WGP; MI455X has >= 64 WGPs).
// ---------------------------------------------------------------------------
__device__ __forceinline__ void grid_sync(unsigned* cnt, unsigned* gen, unsigned nb) {
  __threadfence();
  __syncthreads();
  if (threadIdx.x == 0) {
    unsigned g = __hip_atomic_load(gen, __ATOMIC_RELAXED, __HIP_MEMORY_SCOPE_AGENT);
    if (__hip_atomic_fetch_add(cnt, 1u, __ATOMIC_ACQ_REL, __HIP_MEMORY_SCOPE_AGENT) == nb - 1u) {
      __hip_atomic_store(cnt, 0u, __ATOMIC_RELAXED, __HIP_MEMORY_SCOPE_AGENT);
      __hip_atomic_fetch_add(gen, 1u, __ATOMIC_RELEASE, __HIP_MEMORY_SCOPE_AGENT);
    } else {
      while (__hip_atomic_load(gen, __ATOMIC_ACQUIRE, __HIP_MEMORY_SCOPE_AGENT) == g)
        __builtin_amdgcn_s_sleep(2);
    }
  }
  __syncthreads();
}

// ---------------------------------------------------------------------------
// Pack a row-major f32 matrix M[R x C] into fragment-major bf16 hi/lo arrays.
// Fragment = (16 rows) x (32 k); element order inside a fragment matches the
// CDNA5 16-bit A/B WMMA layout (ISA 7.12.2):
//   lane&15 -> row within tile, lane>>4 -> k-half (kh = 0/8)
//   j<8 -> k = kh+j ; j>=8 -> k = 16+kh+(j-8)
// Each lane's 16 values land contiguously (32B) => GEMM loads are coalesced.
// ---------------------------------------------------------------------------
__global__ __launch_bounds__(256) void pack_bf16_kernel(
    const float* __restrict__ M, __bf16* __restrict__ hi, __bf16* __restrict__ lo,
    int C, int KT, size_t total)
{
  size_t o = (size_t)blockIdx.x * 256 + threadIdx.x;
  if (o >= total) return;
  const int    j    = (int)(o & 15);
  const int    lane = (int)((o >> 4) & 31);
  const size_t f    = o >> 9;               // fragment index = tr*KT + kc
  const int    kc   = (int)(f % (size_t)KT);
  const int    tr   = (int)(f / (size_t)KT);
  const int    row  = tr * 16 + (lane & 15);
  const int    kh   = (lane >> 4) * 8;
  const int    k    = kc * 32 + ((j < 8) ? (kh + j) : (16 + kh + (j - 8)));
  const float  v    = M[(size_t)row * C + k];
  const __bf16 h    = (__bf16)v;
  hi[o] = h;
  lo[o] = (__bf16)(v - (float)h);
}

// ---------------------------------------------------------------------------
// GX[T x N] = A @ B^T + (b1+b2), from pre-packed bf16 fragments.
// Wave tile 16(m) x 64(n), block = 8 waves stacked in m -> 128 x 64 tile.
// Inner loop: 10 x 32B loads + 12 WMMA; bf16x3 (drop lo*lo term).
// ---------------------------------------------------------------------------
__global__ __launch_bounds__(256) void gemm_gx_kernel(
    const __bf16* __restrict__ Ahi, const __bf16* __restrict__ Alo,
    const __bf16* __restrict__ Bhi, const __bf16* __restrict__ Blo,
    const float* __restrict__ b1, const float* __restrict__ b2,
    float* __restrict__ GX, int KT, int N)
{
  const int wave   = threadIdx.x >> 5;
  const int lane   = threadIdx.x & 31;
  const int l16    = lane & 15;
  const int mtile  = blockIdx.x * 8 + wave;
  const int ntile0 = blockIdx.y * 4;

  v8f acc[4] = {};

  size_t af = (size_t)mtile * KT * 512 + (size_t)lane * 16;
  size_t bfr[4];
#pragma unroll
  for (int nt = 0; nt < 4; ++nt)
    bfr[nt] = (size_t)(ntile0 + nt) * KT * 512 + (size_t)lane * 16;

  for (int kc = 0; kc < KT; ++kc) {
    const v16bf ah = *(const v16bf*)(Ahi + af);
    const v16bf al = *(const v16bf*)(Alo + af);
    af += 512;
#pragma unroll
    for (int nt = 0; nt < 4; ++nt) {
      const v16bf bh = *(const v16bf*)(Bhi + bfr[nt]);
      const v16bf bl = *(const v16bf*)(Blo + bfr[nt]);
      bfr[nt] += 512;
      v8f c = acc[nt];
      c = __builtin_amdgcn_wmma_f32_16x16x32_bf16(false, al, false, bh, (short)0, c, false, false);
      c = __builtin_amdgcn_wmma_f32_16x16x32_bf16(false, ah, false, bl, (short)0, c, false, false);
      c = __builtin_amdgcn_wmma_f32_16x16x32_bf16(false, ah, false, bh, (short)0, c, false, false);
      acc[nt] = c;
    }
  }

  // C/D layout: lane<16 -> rows 0-7 of tile, lane>=16 -> rows 8-15.
  const int mbase = mtile * 16 + (lane >> 4) * 8;
#pragma unroll
  for (int nt = 0; nt < 4; ++nt) {
    const int n = (ntile0 + nt) * 16 + l16;
    const float bias = b1[n] + b2[n];
#pragma unroll
    for (int r = 0; r < 8; ++r)
      GX[(size_t)(mbase + r) * N + n] = acc[nt][r] + bias;
  }
}

// ---------------------------------------------------------------------------
// Persistent LSTM scan. Block b owns hidden units j0=b*16 .. j0+15, i.e. the
// 64 gate rows {g*H + j0 + u}. Those rows (64 x 1024 f32 = 256KB) live in LDS
// for all 8192 steps (CDNA5: 320KB LDS/WGP). Each step:
//   h -> LDS, 64 dots (4 lanes/row, q-blocked k with float4 LDS reads),
//   shfl_xor reduce, pointwise gates on 16 lanes, device barrier.
// ---------------------------------------------------------------------------
__global__ __launch_bounds__(256) void lstm_rec_kernel(
    const float* __restrict__ GX,    // T x 4096 (precomputed input gates)
    const float* __restrict__ Whh,   // 4096 x 1024
    const float* __restrict__ h0,    // 1024 (this layer's slice)
    const float* __restrict__ c0,    // 1024
    float* __restrict__ Y,           // T x 1024
    float* __restrict__ hbuf,        // 2 x 1024 (double buffer, global)
    unsigned* __restrict__ syncv,    // [0]=count, [1]=generation
    float* __restrict__ hT, float* __restrict__ cT)
{
  extern __shared__ float smem[];
  float* whh_s = smem;                 // 64 * SROW
  float* h_s   = smem + 64 * SROW;     // 1024
  float* g_s   = h_s + Hdim;           // 64

  const int tid = threadIdx.x;
  const int j0  = blockIdx.x * 16;

  // Load our 64 gate rows of Whh into LDS once (padded stride SROW).
  for (int idx = tid; idx < 64 * Hdim; idx += 256) {
    int lr = idx >> 10;                // local row 0..63
    int k  = idx & (Hdim - 1);
    int gate = lr >> 4, u = lr & 15;
    whh_s[lr * SROW + k] = Whh[(size_t)(gate * Hdim + j0 + u) * Hdim + k];
  }

  float c = 0.0f, hlast = 0.0f;
  if (tid < 16) {
    c = c0[j0 + tid];
    hbuf[j0 + tid] = h0[j0 + tid];     // publish h0 slice into buffer 0
  }
  grid_sync(syncv, syncv + 1, NBLK);   // full h0 visible everywhere

  const int r = tid >> 2;              // local gate row 0..63
  const int q = tid & 3;               // quarter of the K dimension
  const float4* wr4 = (const float4*)(whh_s + r * SROW + q * 256);
  const float4* hq4 = (const float4*)(h_s + q * 256);

  for (int t = 0; t < Tseq; ++t) {
    // Stage h_t into LDS (coalesced).
    const float* hin = hbuf + (t & 1) * Hdim;
#pragma unroll
    for (int i = 0; i < 4; ++i) h_s[tid + i * 256] = hin[tid + i * 256];
    __syncthreads();

    // dot(Whh[row], h) over this thread's 256-wide quarter, float4 LDS reads.
    float s = 0.0f;
#pragma unroll 8
    for (int i = 0; i < 64; ++i) {
      const float4 w4 = wr4[i];
      const float4 h4 = hq4[i];
      s = fmaf(w4.x, h4.x, s);
      s = fmaf(w4.y, h4.y, s);
      s = fmaf(w4.z, h4.z, s);
      s = fmaf(w4.w, h4.w, s);
    }
    s += __shfl_xor(s, 1, 32);
    s += __shfl_xor(s, 2, 32);
    if (q == 0) g_s[r] = s;
    __syncthreads();

    if (tid < 16) {
      const float* gxr = GX + (size_t)t * GATES;
      const int u = tid;
      float iv = sigf (g_s[u]      + gxr[             j0 + u]);
      float fv = sigf (g_s[16 + u] + gxr[    Hdim  +  j0 + u]);
      float gv = tanhf(g_s[32 + u] + gxr[2 * Hdim  +  j0 + u]);
      float ov = sigf (g_s[48 + u] + gxr[3 * Hdim  +  j0 + u]);
      c = fv * c + iv * gv;
      float h = ov * tanhf(c);
      Y[(size_t)t * Hdim + j0 + u] = h;
      hbuf[((t + 1) & 1) * Hdim + j0 + u] = h;
      hlast = h;
    }
    grid_sync(syncv, syncv + 1, NBLK);
  }

  if (tid < 16) {
    hT[j0 + tid] = hlast;
    cT[j0 + tid] = c;
  }
}

// ---------------------------------------------------------------------------
// out[t] = dot(Y[t,:], Wlin[0,:]) + blin[0]   (one wave per row)
// ---------------------------------------------------------------------------
__global__ __launch_bounds__(256) void linear_kernel(
    const float* __restrict__ Y, const float* __restrict__ Wlin,
    const float* __restrict__ blin, float* __restrict__ out)
{
  const int row  = blockIdx.x * 8 + (threadIdx.x >> 5);
  const int lane = threadIdx.x & 31;
  float s = 0.0f;
#pragma unroll 4
  for (int k = lane; k < Hdim; k += 32)
    s = fmaf(Y[(size_t)row * Hdim + k], Wlin[k], s);
#pragma unroll
  for (int off = 16; off; off >>= 1) s += __shfl_xor(s, off, 32);
  if (lane == 0) out[row] = s + blin[0];
}

// ---------------------------------------------------------------------------
extern "C" void kernel_launch(void* const* d_in, const int* in_sizes, int n_in,
                              void* d_out, int out_size, void* d_ws, size_t ws_size,
                              hipStream_t stream)
{
  (void)in_sizes; (void)n_in; (void)out_size; (void)ws_size;
  const float* inputs = (const float*)d_in[0];
  const float* h0     = (const float*)d_in[1];   // (2,1,1024) flat
  const float* c0     = (const float*)d_in[2];
  const float* Wih0   = (const float*)d_in[3];
  const float* Whh0   = (const float*)d_in[4];
  const float* bih0   = (const float*)d_in[5];
  const float* bhh0   = (const float*)d_in[6];
  const float* Wih1   = (const float*)d_in[7];
  const float* Whh1   = (const float*)d_in[8];
  const float* bih1   = (const float*)d_in[9];
  const float* bhh1   = (const float*)d_in[10];
  const float* Wlin   = (const float*)d_in[11];
  const float* blin   = (const float*)d_in[12];
  float* out = (float*)d_out;   // layout: out(8192) | hT(2*1024) | cT(2*1024)

  // --- workspace layout (bytes) ---
  char* ws = (char*)d_ws;
  float*  gx   = (float*)(ws);                            // 8192*4096 f32 = 128MB
  float*  y0   = (float*)(ws + (size_t)134217728);        // 32MB
  float*  y1   = (float*)(ws + (size_t)167772160);        // 32MB
  __bf16* axhi = (__bf16*)(ws + (size_t)201326592);       // <=8M elems = 16MB
  __bf16* axlo = (__bf16*)(ws + (size_t)218103808);       // 16MB
  __bf16* bwhi = (__bf16*)(ws + (size_t)234881024);       // <=4M elems = 8MB
  __bf16* bwlo = (__bf16*)(ws + (size_t)243269632);       // 8MB
  float*  hbuf = (float*)(ws + (size_t)251658240);        // 2*1024 f32
  unsigned* syncv = (unsigned*)(ws + (size_t)251666432);  // 2 u32

  hipMemsetAsync(syncv, 0, 2 * sizeof(unsigned), stream);

  const dim3   blk(256);
  const dim3   ggrid(Tseq / 128, GATES / 64);
  const size_t smem = (size_t)(64 * SROW + Hdim + 64) * sizeof(float);

  // ---------------- Layer 0 ----------------
  {
    const size_t nA = (size_t)Tseq * Fdim;     // 2M
    const size_t nB = (size_t)GATES * Fdim;    // 1M
    pack_bf16_kernel<<<(unsigned)(nA / 256), blk, 0, stream>>>(inputs, axhi, axlo, Fdim, Fdim / 32, nA);
    pack_bf16_kernel<<<(unsigned)(nB / 256), blk, 0, stream>>>(Wih0,   bwhi, bwlo, Fdim, Fdim / 32, nB);
    gemm_gx_kernel<<<ggrid, blk, 0, stream>>>(axhi, axlo, bwhi, bwlo, bih0, bhh0, gx, Fdim / 32, GATES);
    lstm_rec_kernel<<<NBLK, blk, smem, stream>>>(gx, Whh0, h0, c0, y0, hbuf, syncv,
                                                 out + Tseq, out + Tseq + 2 * Hdim);
  }
  // ---------------- Layer 1 ----------------
  {
    const size_t nA = (size_t)Tseq * Hdim;     // 8M
    const size_t nB = (size_t)GATES * Hdim;    // 4M
    pack_bf16_kernel<<<(unsigned)(nA / 256), blk, 0, stream>>>(y0,   axhi, axlo, Hdim, Hdim / 32, nA);
    pack_bf16_kernel<<<(unsigned)(nB / 256), blk, 0, stream>>>(Wih1, bwhi, bwlo, Hdim, Hdim / 32, nB);
    gemm_gx_kernel<<<ggrid, blk, 0, stream>>>(axhi, axlo, bwhi, bwlo, bih1, bhh1, gx, Hdim / 32, GATES);
    lstm_rec_kernel<<<NBLK, blk, smem, stream>>>(gx, Whh1, h0 + Hdim, c0 + Hdim, y1, hbuf, syncv,
                                                 out + Tseq + Hdim, out + Tseq + 2 * Hdim + Hdim);
  }
  // ---------------- Head ----------------
  linear_kernel<<<Tseq / 8, blk, 0, stream>>>(y1, Wlin, blin, out);
}
